// Wave_head_33775622815791
// MI455X (gfx1250) — compile-verified, matrix operand order checked by reading, not run on hardware
//
#include <hip/hip_runtime.h>

#define B_   32
#define CIN  21
#define CH   128
#define LEN  4096
#define RTS  8

typedef __attribute__((ext_vector_type(16))) __bf16      v16bf;
typedef __attribute__((ext_vector_type(8)))  float       v8f;
typedef __attribute__((ext_vector_type(4)))  unsigned int u32x4;
typedef __attribute__((ext_vector_type(8)))  unsigned int u32x8;

static __device__ __forceinline__ float bf2f(unsigned short u) {
  unsigned v = ((unsigned)u) << 16;
  return __builtin_bit_cast(float, v);
}
static __device__ __forceinline__ unsigned short f2bf(float f) {
  unsigned u = __builtin_bit_cast(unsigned, f);
  u = u + 0x7FFFu + ((u >> 16) & 1u);          // round-to-nearest-even
  return (unsigned short)(u >> 16);
}
static __device__ __forceinline__ unsigned pk2(float a, float b) {
  return (unsigned)f2bf(a) | ((unsigned)f2bf(b) << 16);
}

// Load one WMMA 16-bf16 fragment: two contiguous 16B chunks at p and p+16
// (matches ISA A/B layout: lo-lane K{0..7}+K{16..23}, hi-lane K{8..15}+K{24..31}
// when caller offsets p by hi*8).
static __device__ __forceinline__ v16bf frag16(const unsigned short* p) {
  u32x4 lo = *(const u32x4*)p;
  u32x4 hi = *(const u32x4*)(p + 16);
  u32x8 u = {lo.x, lo.y, lo.z, lo.w, hi.x, hi.y, hi.z, hi.w};
  return __builtin_bit_cast(v16bf, u);
}

// ---------------------------------------------------------------------------
// Prep: convert/rearrange weights to bf16.
//   Wsig/Wgate : [layer][o][tap*128 + i]   (K-major for the fused-tap GEMM)
//   Wres/Wskip : [layer][o][i]
// ---------------------------------------------------------------------------
__global__ void wn_prep_weights(const float* __restrict__ sig,
                                const float* __restrict__ gat,
                                const float* __restrict__ res,
                                const float* __restrict__ skp,
                                unsigned short* __restrict__ Wsig,
                                unsigned short* __restrict__ Wgate,
                                unsigned short* __restrict__ Wres,
                                unsigned short* __restrict__ Wskip) {
  int idx = blockIdx.x * blockDim.x + threadIdx.x;
  const int total_big = RTS * CH * 256;          // 262144
  if (idx < total_big) {
    int ii  = idx >> 15;                         // /(128*256)
    int rem = idx & 32767;
    int o   = rem >> 8;
    int k   = rem & 255;
    int tap = k >> 7;
    int i   = k & 127;
    int src = ((ii * CH + o) * CH + i) * 2 + tap;
    Wsig[idx]  = f2bf(sig[src]);
    Wgate[idx] = f2bf(gat[src]);
  }
  const int total_small = RTS * CH * CH;         // 131072
  if (idx < total_small) {
    Wres[idx]  = f2bf(res[idx]);
    Wskip[idx] = f2bf(skp[idx]);
  }
}

// ---------------------------------------------------------------------------
// Input projection: h0[b][t][c] = sum_i w0[c][i]*x[b][i][t] + b0[c]  (bf16 out)
// K=21 is tiny (<1% of FLOPs) -> VALU in f32.
// ---------------------------------------------------------------------------
__global__ void wn_proj(const float* __restrict__ x, const float* __restrict__ w0,
                        const float* __restrict__ b0, unsigned short* __restrict__ h0) {
  __shared__ float xs[CIN * 64];
  const int b  = blockIdx.x >> 6;
  const int t0 = (blockIdx.x & 63) << 6;
  for (int e = threadIdx.x; e < CIN * 64; e += 256) {
    int i = e >> 6, n = e & 63;
    xs[e] = x[(size_t)(b * CIN + i) * LEN + t0 + n];
  }
  __syncthreads();
  for (int o = threadIdx.x; o < 64 * CH; o += 256) {
    int n = o >> 7, c = o & 127;
    float acc = b0[c];
#pragma unroll
    for (int i = 0; i < CIN; ++i) acc += w0[c * CIN + i] * xs[(i << 6) + n];
    h0[((size_t)(b * LEN + t0 + n) << 7) + c] = f2bf(acc);
  }
}

// ---------------------------------------------------------------------------
// One WaveNet layer, fully fused. Tile: M=128 channels x N=64 timesteps.
// 8 waves; wave w owns output rows [16w, 16w+16).
// ---------------------------------------------------------------------------
#define XS_STRIDE 264   // 256 + 8 (16B-aligned, bank-spread)
#define GS_STRIDE 136   // 128 + 8

__global__ void __launch_bounds__(256)
wn_layer(const unsigned short* __restrict__ hprev,
         unsigned short* __restrict__ hnext,
         const unsigned short* __restrict__ Wsig,
         const unsigned short* __restrict__ Wgate,
         const unsigned short* __restrict__ Wres,
         const unsigned short* __restrict__ Wskip,
         const float* __restrict__ condS,
         const float* __restrict__ condG,
         const float* __restrict__ z,
         float* __restrict__ out_h,
         float* __restrict__ out_skip,
         int d, int flags) {
  __shared__ unsigned short Xs[64 * XS_STRIDE];   // K=256 x N=64 (taps fused)
  __shared__ unsigned short Gs[64 * GS_STRIDE];   // K=128 x N=64 gated acts

  const int tid  = threadIdx.x;
  const int lane = tid & 31;
  const int wave = tid >> 5;            // M-block
  const int hi   = lane >> 4;
  const int n16  = lane & 15;
  const int b    = blockIdx.x >> 6;
  const int t0   = (blockIdx.x & 63) << 6;
  const bool first = flags & 1;
  const bool last  = flags & 2;

  // ---- stage X tile into LDS: row n holds [h[:,t-d-1] ; h[:,t-1]] ----
  for (int chunk = tid; chunk < 64 * 32; chunk += 256) {
    int n   = chunk >> 5;
    int j   = chunk & 31;
    int tap = j >> 4;                               // 0 -> t-d-1, 1 -> t-1
    int ci  = (j & 15) << 3;
    int tsrc = t0 + n - (tap == 0 ? (d + 1) : 1);
    u32x4 v = {0u, 0u, 0u, 0u};
    if (tsrc >= 0)
      v = *(const u32x4*)(hprev + (((size_t)(b * LEN + tsrc)) << 7) + ci);
    *(u32x4*)(&Xs[n * XS_STRIDE + (tap << 7) + ci]) = v;
  }
  __syncthreads();

  // ---- GEMM1: signal/gate = W[128x256] * X[256x64] ----
  v8f accS[4], accG[4];
#pragma unroll
  for (int nt = 0; nt < 4; ++nt) {
    accS[nt] = (v8f){0.f,0.f,0.f,0.f,0.f,0.f,0.f,0.f};
    accG[nt] = (v8f){0.f,0.f,0.f,0.f,0.f,0.f,0.f,0.f};
  }
  const int row = wave * 16 + n16;      // A-matrix row (all lanes: M = lane&15)
  const int grp = hi << 3;              // hi-lane half picks K{8..15,24..31}
#pragma unroll
  for (int kt = 0; kt < 8; ++kt) {
    v16bf aS = frag16(Wsig  + row * 256 + kt * 32 + grp);
    v16bf aG = frag16(Wgate + row * 256 + kt * 32 + grp);
#pragma unroll
    for (int nt = 0; nt < 4; ++nt) {
      v16bf bx = frag16(&Xs[(nt * 16 + n16) * XS_STRIDE + kt * 32 + grp]);
      accS[nt] = __builtin_amdgcn_wmma_f32_16x16x32_bf16(false, aS, false, bx,
                   (short)0, accS[nt], false, false);
      accG[nt] = __builtin_amdgcn_wmma_f32_16x16x32_bf16(false, aG, false, bx,
                   (short)0, accG[nt], false, false);
    }
  }

  // ---- conditioning + gated activation; park g (bf16) in LDS for GEMM2 ----
  const int cbase = wave * 16 + (hi << 3);   // 8 contiguous channels per lane
  float cS[8], cG[8];
#pragma unroll
  for (int r = 0; r < 8; ++r) { cS[r] = condS[cbase + r]; cG[r] = condG[cbase + r]; }

  v8f g[4];
#pragma unroll
  for (int nt = 0; nt < 4; ++nt) {
    float zt = z[(size_t)b * LEN + t0 + nt * 16 + n16];
#pragma unroll
    for (int r = 0; r < 8; ++r) {
      float s  = accS[nt][r] + cS[r] * zt;
      float gt = accG[nt][r] + cG[r] * zt;
      g[nt][r] = s * __builtin_amdgcn_rcpf(1.f + __expf(-gt));
    }
    u32x4 pk;
    pk.x = pk2(g[nt][0], g[nt][1]);
    pk.y = pk2(g[nt][2], g[nt][3]);
    pk.z = pk2(g[nt][4], g[nt][5]);
    pk.w = pk2(g[nt][6], g[nt][7]);
    *(u32x4*)(&Gs[(nt * 16 + n16) * GS_STRIDE + cbase]) = pk;
  }
  __syncthreads();

  // ---- GEMM2: res/skip = W[128x128] * G[128x64] ----
  v8f accR[4], accK[4];
#pragma unroll
  for (int nt = 0; nt < 4; ++nt) {
    accR[nt] = (v8f){0.f,0.f,0.f,0.f,0.f,0.f,0.f,0.f};
    accK[nt] = (v8f){0.f,0.f,0.f,0.f,0.f,0.f,0.f,0.f};
  }
#pragma unroll
  for (int kt = 0; kt < 4; ++kt) {
    v16bf aR = frag16(Wres  + row * 128 + kt * 32 + grp);
    v16bf aK = frag16(Wskip + row * 128 + kt * 32 + grp);
#pragma unroll
    for (int nt = 0; nt < 4; ++nt) {
      v16bf bg = frag16(&Gs[(nt * 16 + n16) * GS_STRIDE + kt * 32 + grp]);
      accR[nt] = __builtin_amdgcn_wmma_f32_16x16x32_bf16(false, aR, false, bg,
                   (short)0, accR[nt], false, false);
      accK[nt] = __builtin_amdgcn_wmma_f32_16x16x32_bf16(false, aK, false, bg,
                   (short)0, accK[nt], false, false);
    }
  }

  // ---- epilogue: skip accumulate, residual update, h write ----
#pragma unroll
  for (int nt = 0; nt < 4; ++nt) {
    const int t = t0 + nt * 16 + n16;
    const size_t srow = (size_t)(b * CH + cbase) * LEN + t;  // [b][c][t] f32
#pragma unroll
    for (int r = 0; r < 8; ++r) {
      size_t idx = srow + (size_t)r * LEN;
      out_skip[idx] = first ? accK[nt][r] : (out_skip[idx] + accK[nt][r]);
    }
    float hv[8];
    if (first) {   // h = orig + res ; orig == hprev (projection output)
      const unsigned short* op = hprev + (((size_t)(b * LEN + t)) << 7) + cbase;
#pragma unroll
      for (int r = 0; r < 8; ++r) hv[r] = bf2f(op[r]) + accR[nt][r];
    } else {       // h = res + g
#pragma unroll
      for (int r = 0; r < 8; ++r) hv[r] = accR[nt][r] + g[nt][r];
    }
    if (last) {
#pragma unroll
      for (int r = 0; r < 8; ++r) out_h[srow + (size_t)r * LEN] = hv[r];
    } else {
      u32x4 pk;
      pk.x = pk2(hv[0], hv[1]);
      pk.y = pk2(hv[2], hv[3]);
      pk.z = pk2(hv[4], hv[5]);
      pk.w = pk2(hv[6], hv[7]);
      *(u32x4*)(hnext + (((size_t)(b * LEN + t)) << 7) + cbase) = pk;
    }
  }
}

// ---------------------------------------------------------------------------
extern "C" void kernel_launch(void* const* d_in, const int* in_sizes, int n_in,
                              void* d_out, int out_size, void* d_ws, size_t ws_size,
                              hipStream_t stream) {
  const float* x           = (const float*)d_in[0];
  const float* z           = (const float*)d_in[1];
  const float* w0          = (const float*)d_in[2];
  const float* b0          = (const float*)d_in[3];
  const float* signal_w    = (const float*)d_in[4];
  const float* gate_w      = (const float*)d_in[5];
  const float* cond_signal = (const float*)d_in[6];
  const float* cond_gate   = (const float*)d_in[7];
  const float* res_w       = (const float*)d_in[8];
  const float* skip_w      = (const float*)d_in[9];

  float* out_h    = (float*)d_out;
  float* out_skip = out_h + (size_t)B_ * CH * LEN;

  const size_t hElems = (size_t)B_ * LEN * CH;     // 16,777,216
  unsigned short* ws    = (unsigned short*)d_ws;
  unsigned short* hA    = ws;
  unsigned short* hB    = hA + hElems;
  unsigned short* Wsig  = hB + hElems;
  unsigned short* Wgate = Wsig  + (size_t)RTS * CH * 256;
  unsigned short* Wres  = Wgate + (size_t)RTS * CH * 256;
  unsigned short* Wskip = Wres  + (size_t)RTS * CH * CH;

  wn_prep_weights<<<1024, 256, 0, stream>>>(signal_w, gate_w, res_w, skip_w,
                                            Wsig, Wgate, Wres, Wskip);
  const int nTiles = B_ * (LEN / 64);   // 2048 workgroups
  wn_proj<<<nTiles, 256, 0, stream>>>(x, w0, b0, hA);

  unsigned short* hp = hA;
  unsigned short* hn = hB;
  for (int ii = 0; ii < RTS; ++ii) {
    const int flags = (ii == 0 ? 1 : 0) | (ii == RTS - 1 ? 2 : 0);
    wn_layer<<<nTiles, 256, 0, stream>>>(
        hp, hn,
        Wsig + (size_t)ii * CH * 256, Wgate + (size_t)ii * CH * 256,
        Wres + (size_t)ii * CH * CH,  Wskip + (size_t)ii * CH * CH,
        cond_signal + ii * CH, cond_gate + ii * CH, z,
        out_h, out_skip, 1 << ii, flags);
    unsigned short* tmp = hp; hp = hn; hn = tmp;
  }
}
